// ASR_Decoder_58136677319026
// MI455X (gfx1250) — matrix operand order repaired and checked by reading, not compiled, not run on hardware
//
#include <hip/hip_runtime.h>
#include <hip/hip_bf16.h>

// ---------------- problem constants ----------------
#define B_   32
#define T_   256
#define K_   512        // encoder frames
#define V_   10000      // vocab
#define EMB_ 64
#define H_   256
#define DM_  512
#define G4_  1024       // 4*H
#define SOS_ 1
#define EOS_ 2
#define PAD_ 0
#define TB_  (T_*B_)    // 8192

typedef __bf16 bf16_t;
typedef __bf16 v16bf __attribute__((ext_vector_type(16)));
typedef float  v8f   __attribute__((ext_vector_type(8)));
typedef unsigned int u32x4 __attribute__((ext_vector_type(4)));
typedef int          i32x4 __attribute__((ext_vector_type(4)));

// ---------------- workspace layout (bytes) ----------------
#define OFF_PRE0  ((size_t)0)
#define SZ_PRE0   ((size_t)TB_*G4_*4)        // f32 gate0 preact (emb part + b0)
#define OFF_WOUT  (OFF_PRE0 + SZ_PRE0)
#define SZ_WOUT   ((size_t)V_*G4_*2)         // bf16 W_out
#define OFF_W0C   (OFF_WOUT + SZ_WOUT)
#define SZ_W0C    ((size_t)G4_*DM_*2)        // bf16 ctx-part of W_ih0
#define OFF_W1    (OFF_W0C + SZ_W0C)
#define SZ_W1     ((size_t)G4_*H_*2)
#define OFF_WP    (OFF_W1 + SZ_W1)
#define SZ_WP     ((size_t)DM_*H_*2)
#define OFF_AB    (OFF_WP + SZ_WP)
#define SZ_AB     ((size_t)TB_*G4_*2)        // bf16 concat(proj,ctx) rows t*B+b
#define OFF_G     (OFF_AB + SZ_AB)
#define SZ_G      ((size_t)B_*G4_*4)         // per-step gate buffer
#define OFF_PROJ  (OFF_G + SZ_G)
#define SZ_PROJ   ((size_t)B_*DM_*4)
#define OFF_SCORE (OFF_PROJ + SZ_PROJ)
#define SZ_SCORE  ((size_t)B_*K_*4)

// ---------------- CDNA5 async-to-LDS support (probe via __has_builtin) ----------------
#if __has_builtin(__builtin_amdgcn_global_load_async_to_lds_b128)
#define ASYNC_LDS 1
#else
#define ASYNC_LDS 0
#endif

#if ASYNC_LDS
typedef __attribute__((address_space(1))) i32x4* gp1_t;   // global int4*
typedef __attribute__((address_space(3))) i32x4* lp3_t;   // LDS int4*

__device__ __forceinline__ void async_cp16(const void* g, void* l) {
  // GLOBAL_LOAD_ASYNC_TO_LDS_B128: LDS[l] = MEM[g]  (tracked by ASYNCcnt)
  __builtin_amdgcn_global_load_async_to_lds_b128(
      (gp1_t)(uintptr_t)g, (lp3_t)(uintptr_t)l, /*offset=*/0, /*cpol=*/0);
}
__device__ __forceinline__ void wait_async_le2() {
#if __has_builtin(__builtin_amdgcn_s_wait_asynccnt)
  __builtin_amdgcn_s_wait_asynccnt(2);
#else
  asm volatile("s_wait_asynccnt 0x2" ::: "memory");
#endif
}
__device__ __forceinline__ void wait_async_le0() {
#if __has_builtin(__builtin_amdgcn_s_wait_asynccnt)
  __builtin_amdgcn_s_wait_asynccnt(0);
#else
  asm volatile("s_wait_asynccnt 0x0" ::: "memory");
#endif
}
#endif

// ---------------- WMMA helpers ----------------
__device__ __forceinline__ v8f wmma_bf16(v16bf a, v16bf b, v8f c) {
  return __builtin_amdgcn_wmma_f32_16x16x32_bf16(false, a, false, b, (short)0, c,
                                                 false, false);
}

// A fragment 16x32 bf16: lanes 0-15 rows 0-15, K 0..15;
// lanes 16-31 rows 0-15, K 16..31. Two aligned 16B loads per lane.
__device__ __forceinline__ v16bf load_frag_a(const bf16_t* base, int stride, int lane) {
  int r  = lane & 15;
  int ko = (lane >> 4) << 4;                // 0 or 16
  const bf16_t* p = base + r * stride + ko;
  union { v16bf v; u32x4 q[2]; } u;
  u.q[0] = *(const u32x4*)(p);              // K = ko..ko+7
  u.q[1] = *(const u32x4*)(p + 8);          // K = ko+8..ko+15
  return u.v;
}

// B fragment 32x16 bf16: lane n = lane&15 (column of C), lanes 0-15 K 0..15,
// lanes 16-31 K 16..31. Row-major [N,K] weights -> contiguous 32B per lane.
__device__ __forceinline__ v16bf load_frag_b(const bf16_t* base, int stride, int lane) {
  int n  = lane & 15;
  int ko = (lane >> 4) << 4;                // 0 or 16
  const bf16_t* p = base + n * stride + ko;
  union { v16bf v; u32x4 q[2]; } u;
  u.q[0] = *(const u32x4*)(p);
  u.q[1] = *(const u32x4*)(p + 8);
  return u.v;
}

__device__ __forceinline__ float sigf(float x) { return 1.f / (1.f + __expf(-x)); }

// ---------------- kernel 1: weight conversion to bf16 ----------------
__global__ __launch_bounds__(256) void cvt_weights(
    const float* __restrict__ Wih0, const float* __restrict__ Wih1,
    const float* __restrict__ Wproj, const float* __restrict__ Wout,
    bf16_t* __restrict__ w0c, bf16_t* __restrict__ w1,
    bf16_t* __restrict__ wp,  bf16_t* __restrict__ wo) {
  const long n0 = (long)V_ * G4_;
  const long n1 = (long)G4_ * DM_;
  const long n2 = (long)G4_ * H_;
  const long n3 = (long)DM_ * H_;
  const long tot = n0 + n1 + n2 + n3;
  for (long i = (long)blockIdx.x * blockDim.x + threadIdx.x; i < tot;
       i += (long)gridDim.x * blockDim.x) {
    if (i < n0) {
      wo[i] = (bf16_t)Wout[i];
    } else if (i < n0 + n1) {
      long j = i - n0;
      long row = j / DM_, col = j - row * DM_;
      w0c[j] = (bf16_t)Wih0[row * (EMB_ + DM_) + EMB_ + col];  // skip emb cols
    } else if (i < n0 + n1 + n2) {
      long j = i - n0 - n1;
      w1[j] = (bf16_t)Wih1[j];
    } else {
      long j = i - n0 - n1 - n2;
      wp[j] = (bf16_t)Wproj[j];
    }
  }
}

// ---------------- kernel 2: token prep + embedding part of gate0 ----------------
__global__ __launch_bounds__(256) void pre0_kernel(
    const int* __restrict__ dec, const float* __restrict__ emb,
    const float* __restrict__ Wih0, const float* __restrict__ b0,
    float* __restrict__ pre0) {
  long flat = (long)blockIdx.x * blockDim.x + threadIdx.x;
  if (flat >= (long)TB_ * G4_) return;
  int j   = (int)(flat & (G4_ - 1));
  int row = (int)(flat >> 10);              // t*B + b
  int t = row >> 5;
  int b = row & 31;
  int tok = (t == 0) ? SOS_ : dec[b * T_ + (t - 1)];
  if (tok == EOS_) tok = PAD_;
  const float4* e4 = (const float4*)(emb + (long)tok * EMB_);
  const float4* w4 = (const float4*)(Wih0 + (long)j * (EMB_ + DM_));
  float acc = b0[j];
#pragma unroll
  for (int k = 0; k < EMB_ / 4; ++k) {
    float4 e = e4[k], w = w4[k];
    acc += e.x * w.x + e.y * w.y + e.z * w.z + e.w * w.w;
  }
  pre0[flat] = acc;
}

// ---------------- kernel 3: sequential recurrence (1 WG, 16 waves, WMMA) ----------------
__global__ __launch_bounds__(512) void decoder_recurrence(
    const float* __restrict__ enc, const float* __restrict__ pre0,
    const bf16_t* __restrict__ w0c, const bf16_t* __restrict__ w1,
    const float* __restrict__ b1v, const bf16_t* __restrict__ wp,
    const float* __restrict__ bp, float* __restrict__ gbuf,
    float* __restrict__ projf, float* __restrict__ score,
    bf16_t* __restrict__ AB) {
  __shared__ __align__(16) bf16_t ctx_bf[B_][DM_];   // 32 KB
  __shared__ __align__(16) bf16_t h_bf[B_][H_];      // 16 KB
  const int tid  = threadIdx.x;
  const int lane = tid & 31;
  const int wave = tid >> 5;                          // 0..15

  for (int i = tid; i < B_ * DM_; i += 512) ((bf16_t*)ctx_bf)[i] = (bf16_t)0.f;
  __syncthreads();

  for (int t = 0; t < T_; ++t) {
    const float* pre_t = pre0 + (long)t * B_ * G4_;

    // ---- gate0: g[32,1024] = ctx[32,512] @ W0c^T + pre_t  (128 tiles, 8/wave)
    // software-pipelined K loop: next fragments load while current WMMA runs
    for (int i = 0; i < 8; ++i) {
      int tile = wave * 8 + i;
      int mt = tile >> 6, nt = tile & 63;
      const bf16_t* abase = &ctx_bf[mt * 16][0];
      const bf16_t* bbase = w0c + (long)(nt * 16) * DM_;
      v16bf a  = load_frag_a(abase, DM_, lane);
      v16bf bf = load_frag_b(bbase, DM_, lane);
      v8f acc = {};
#pragma unroll 4
      for (int kk = 32; kk <= DM_; kk += 32) {
        int kn = (kk < DM_) ? kk : DM_ - 32;
        v16bf an = load_frag_a(abase + kn, DM_, lane);
        v16bf bn = load_frag_b(bbase + kn, DM_, lane);
        acc = wmma_bf16(a, bf, acc);
        a = an; bf = bn;
      }
      int n  = nt * 16 + (lane & 15);
      int m0 = mt * 16 + ((lane >> 4) << 3);
#pragma unroll
      for (int r = 0; r < 8; ++r) {
        int m = m0 + r;
        gbuf[m * G4_ + n] = acc[r] + pre_t[m * G4_ + n];
      }
    }
    __threadfence_block();
    __syncthreads();

    // ---- LSTM0 elementwise (c_prev = 0 so forget gate drops out)
    for (int i = tid; i < B_ * H_; i += 512) {
      int b = i >> 8, n = i & (H_ - 1);
      float gi = gbuf[b * G4_ + n];
      float gg = gbuf[b * G4_ + 2 * H_ + n];
      float go = gbuf[b * G4_ + 3 * H_ + n];
      float c  = sigf(gi) * tanhf(gg);
      h_bf[b][n] = (bf16_t)(sigf(go) * tanhf(c));
    }
    __syncthreads();

    // ---- gate1: g[32,1024] = h[32,256] @ W1^T + b1
    for (int i = 0; i < 8; ++i) {
      int tile = wave * 8 + i;
      int mt = tile >> 6, nt = tile & 63;
      const bf16_t* abase = &h_bf[mt * 16][0];
      const bf16_t* bbase = w1 + (long)(nt * 16) * H_;
      v16bf a  = load_frag_a(abase, H_, lane);
      v16bf bf = load_frag_b(bbase, H_, lane);
      v8f acc = {};
#pragma unroll
      for (int kk = 32; kk <= H_; kk += 32) {
        int kn = (kk < H_) ? kk : H_ - 32;
        v16bf an = load_frag_a(abase + kn, H_, lane);
        v16bf bn = load_frag_b(bbase + kn, H_, lane);
        acc = wmma_bf16(a, bf, acc);
        a = an; bf = bn;
      }
      int n  = nt * 16 + (lane & 15);
      int m0 = mt * 16 + ((lane >> 4) << 3);
      float bias = b1v[n];
#pragma unroll
      for (int r = 0; r < 8; ++r) gbuf[(m0 + r) * G4_ + n] = acc[r] + bias;
    }
    __threadfence_block();
    __syncthreads();

    // ---- LSTM1 elementwise -> overwrite h_bf
    for (int i = tid; i < B_ * H_; i += 512) {
      int b = i >> 8, n = i & (H_ - 1);
      float gi = gbuf[b * G4_ + n];
      float gg = gbuf[b * G4_ + 2 * H_ + n];
      float go = gbuf[b * G4_ + 3 * H_ + n];
      float c  = sigf(gi) * tanhf(gg);
      h_bf[b][n] = (bf16_t)(sigf(go) * tanhf(c));
    }
    __syncthreads();

    // ---- proj: relu(h @ Wp^T + bp) -> projf (f32) and AB[:, :512] (bf16)
    for (int i = 0; i < 4; ++i) {
      int tile = wave * 4 + i;
      int mt = tile >> 5, nt = tile & 31;
      const bf16_t* abase = &h_bf[mt * 16][0];
      const bf16_t* bbase = wp + (long)(nt * 16) * H_;
      v16bf a  = load_frag_a(abase, H_, lane);
      v16bf bf = load_frag_b(bbase, H_, lane);
      v8f acc = {};
#pragma unroll
      for (int kk = 32; kk <= H_; kk += 32) {
        int kn = (kk < H_) ? kk : H_ - 32;
        v16bf an = load_frag_a(abase + kn, H_, lane);
        v16bf bn = load_frag_b(bbase + kn, H_, lane);
        acc = wmma_bf16(a, bf, acc);
        a = an; bf = bn;
      }
      int n  = nt * 16 + (lane & 15);
      int m0 = mt * 16 + ((lane >> 4) << 3);
      float bias = bp[n];
#pragma unroll
      for (int r = 0; r < 8; ++r) {
        int m = m0 + r;
        float v = fmaxf(acc[r] + bias, 0.f);
        projf[m * DM_ + n] = v;
        AB[(long)(t * B_ + m) * G4_ + n] = (bf16_t)v;
      }
    }
    __threadfence_block();
    __syncthreads();

    // ---- attention scores: score[b,k] = proj[b,:] . enc[b,k,:]
    {
      int b  = tid >> 4;
      int k0 = (tid & 15) * 32;
      const float4* p4 = (const float4*)(projf + b * DM_);
      for (int k = k0; k < k0 + 32; ++k) {
        const float4* e4 = (const float4*)(enc + ((long)b * K_ + k) * DM_);
        float s = 0.f;
#pragma unroll 8
        for (int d = 0; d < DM_ / 4; ++d) {
          float4 e = e4[d], p = p4[d];
          s += e.x * p.x + e.y * p.y + e.z * p.z + e.w * p.w;
        }
        score[b * K_ + k] = s;
      }
    }
    __threadfence_block();
    __syncthreads();

    // ---- softmax over K=512, 2 rows per wave
#pragma unroll
    for (int rr = 0; rr < 2; ++rr) {
      int b = wave * 2 + rr;
      float vals[16];
      float mx = -3.4e38f;
#pragma unroll
      for (int i2 = 0; i2 < 16; ++i2) {
        vals[i2] = score[b * K_ + i2 * 32 + lane];
        mx = fmaxf(mx, vals[i2]);
      }
#pragma unroll
      for (int off = 16; off > 0; off >>= 1) mx = fmaxf(mx, __shfl_xor(mx, off, 32));
      float sum = 0.f;
#pragma unroll
      for (int i2 = 0; i2 < 16; ++i2) { vals[i2] = __expf(vals[i2] - mx); sum += vals[i2]; }
#pragma unroll
      for (int off = 16; off > 0; off >>= 1) sum += __shfl_xor(sum, off, 32);
      float inv = 1.f / sum;
#pragma unroll
      for (int i2 = 0; i2 < 16; ++i2) score[b * K_ + i2 * 32 + lane] = vals[i2] * inv;
    }
    __threadfence_block();
    __syncthreads();

    // ---- context: ctx[b,d] = sum_k attn[b,k] * enc[b,k,d]; feed next step + AB[:,512:]
    {
      int b  = tid >> 4;
      int d0 = (tid & 15) * 32;
      float4 acc[8];
#pragma unroll
      for (int j = 0; j < 8; ++j) acc[j] = make_float4(0.f, 0.f, 0.f, 0.f);
      for (int k = 0; k < K_; ++k) {
        float a = score[b * K_ + k];
        const float4* e4 = (const float4*)(enc + ((long)b * K_ + k) * DM_ + d0);
#pragma unroll
        for (int j = 0; j < 8; ++j) {
          float4 e = e4[j];
          acc[j].x += a * e.x; acc[j].y += a * e.y;
          acc[j].z += a * e.z; acc[j].w += a * e.w;
        }
      }
      bf16_t* abrow = AB + (long)(t * B_ + b) * G4_ + DM_;
#pragma unroll
      for (int j = 0; j < 8; ++j) {
        int d = d0 + j * 4;
        float4 v = acc[j];
        ctx_bf[b][d + 0] = (bf16_t)v.x; ctx_bf[b][d + 1] = (bf16_t)v.y;
        ctx_bf[b][d + 2] = (bf16_t)v.z; ctx_bf[b][d + 3] = (bf16_t)v.w;
        abrow[d + 0] = (bf16_t)v.x; abrow[d + 1] = (bf16_t)v.y;
        abrow[d + 2] = (bf16_t)v.z; abrow[d + 3] = (bf16_t)v.w;
      }
    }
    __syncthreads();
  }
}

// ---------------- kernel 4: output GEMM  out[b,t,v] = AB[t*B+b,:] . Wout[v,:] ----------------
// 64x64 block tile, bf16 LDS panels (row stride 40 halves = conflict-free b128),
// 8 waves x 2 C-tiles, K=1024 in 32 WMMA steps.
// GLOBAL_LOAD_ASYNC_TO_LDS_B128 double-buffers panels through the async engine
// (ASYNCcnt); fallback is synchronous global->VGPR->LDS staging.
__global__ __launch_bounds__(256) void out_gemm(
    const bf16_t* __restrict__ AB, const bf16_t* __restrict__ Wout,
    float* __restrict__ out) {
#if ASYNC_LDS
  __shared__ __align__(16) bf16_t At[2][64][40];
  __shared__ __align__(16) bf16_t Bt[2][64][40];
#else
  __shared__ __align__(16) bf16_t At[1][64][40];
  __shared__ __align__(16) bf16_t Bt[1][64][40];
#endif
  const int tid = threadIdx.x, lane = tid & 31, wave = tid >> 5;
  const int mt    = wave & 3;
  const int npair = wave >> 2;                 // 0 or 1 -> n-tile pair
  const long mbase = (long)blockIdx.y * 64;
  const int  nbase = blockIdx.x * 64;
  const int lrow = tid >> 2;                   // 0..63
  const int lcol = (tid & 3) << 3;             // 0,8,16,24 (elements)
  int vrow = nbase + lrow; if (vrow >= V_) vrow = V_ - 1;   // clamp edge loads
  const bf16_t* aptr = AB + (mbase + lrow) * G4_ + lcol;
  const bf16_t* bptr = Wout + (long)vrow * G4_ + lcol;

  v8f acc0 = {}, acc1 = {};

#if ASYNC_LDS
  async_cp16(aptr, &At[0][lrow][lcol]);
  async_cp16(bptr, &Bt[0][lrow][lcol]);
  int p = 0;
  for (int kk = 0; kk < G4_; kk += 32, p ^= 1) {
    if (kk + 32 < G4_) {
      async_cp16(aptr + kk + 32, &At[p ^ 1][lrow][lcol]);
      async_cp16(bptr + kk + 32, &Bt[p ^ 1][lrow][lcol]);
      wait_async_le2();        // current panel's 2 loads done; next 2 in flight
    } else {
      wait_async_le0();
    }
    __syncthreads();
    v16bf a   = load_frag_a(&At[p][mt * 16][0], 40, lane);
    v16bf bf0 = load_frag_b(&Bt[p][npair * 32][0], 40, lane);
    v16bf bf1 = load_frag_b(&Bt[p][npair * 32 + 16][0], 40, lane);
    acc0 = wmma_bf16(a, bf0, acc0);
    acc1 = wmma_bf16(a, bf1, acc1);
    __syncthreads();           // panel p free before it is re-targeted
  }
#else
  for (int kk = 0; kk < G4_; kk += 32) {
    u32x4 av = *(const u32x4*)(aptr + kk);
    u32x4 bv = *(const u32x4*)(bptr + kk);
    __syncthreads();
    *(u32x4*)(&At[0][lrow][lcol]) = av;
    *(u32x4*)(&Bt[0][lrow][lcol]) = bv;
    __syncthreads();
    v16bf a   = load_frag_a(&At[0][mt * 16][0], 40, lane);
    v16bf bf0 = load_frag_b(&Bt[0][npair * 32][0], 40, lane);
    v16bf bf1 = load_frag_b(&Bt[0][npair * 32 + 16][0], 40, lane);
    acc0 = wmma_bf16(a, bf0, acc0);
    acc1 = wmma_bf16(a, bf1, acc1);
  }
#endif

  int n0 = nbase + npair * 32 + (lane & 15);
  int n1 = n0 + 16;
  long m0 = mbase + mt * 16 + ((lane >> 4) << 3);
#pragma unroll
  for (int r = 0; r < 8; ++r) {
    long m  = m0 + r;            // row = t*B + b
    long tt = m >> 5;
    long bb = m & 31;
    long obase = (bb * T_ + tt) * (long)V_;
    if (n0 < V_) out[obase + n0] = acc0[r];
    if (n1 < V_) out[obase + n1] = acc1[r];
  }
}

// ---------------- host launcher ----------------
extern "C" void kernel_launch(void* const* d_in, const int* in_sizes, int n_in,
                              void* d_out, int out_size, void* d_ws, size_t ws_size,
                              hipStream_t stream) {
  const float* enc   = (const float*)d_in[0];
  const int*   dec   = (const int*)d_in[1];
  const float* emb   = (const float*)d_in[2];
  const float* Wih0  = (const float*)d_in[3];
  const float* b0    = (const float*)d_in[4];
  const float* Wih1  = (const float*)d_in[5];
  const float* b1    = (const float*)d_in[6];
  const float* Wproj = (const float*)d_in[7];
  const float* bproj = (const float*)d_in[8];
  const float* Wout  = (const float*)d_in[9];
  float* out = (float*)d_out;
  char*  ws  = (char*)d_ws;

  float*  pre0   = (float*)(ws + OFF_PRE0);
  bf16_t* woutbf = (bf16_t*)(ws + OFF_WOUT);
  bf16_t* w0cbf  = (bf16_t*)(ws + OFF_W0C);
  bf16_t* w1bf   = (bf16_t*)(ws + OFF_W1);
  bf16_t* wpbf   = (bf16_t*)(ws + OFF_WP);
  bf16_t* ABbf   = (bf16_t*)(ws + OFF_AB);
  float*  gbuf   = (float*)(ws + OFF_G);
  float*  projf  = (float*)(ws + OFF_PROJ);
  float*  scoreb = (float*)(ws + OFF_SCORE);

  hipLaunchKernelGGL(cvt_weights, dim3(2048), dim3(256), 0, stream,
                     Wih0, Wih1, Wproj, Wout, w0cbf, w1bf, wpbf, woutbf);
  hipLaunchKernelGGL(pre0_kernel, dim3((TB_ * G4_) / 256), dim3(256), 0, stream,
                     dec, emb, Wih0, b0, pre0);
  hipLaunchKernelGGL(decoder_recurrence, dim3(1), dim3(512), 0, stream,
                     enc, pre0, w0cbf, w1bf, b1, wpbf, bproj,
                     gbuf, projf, scoreb, ABbf);
  hipLaunchKernelGGL(out_gemm, dim3((V_ + 63) / 64, TB_ / 64), dim3(256), 0, stream,
                     ABbf, woutbf, out);
}
